// VectorQuantizer_3281355014181
// MI455X (gfx1250) — matrix operand order repaired and checked by reading, not compiled, not run on hardware
//
#include <hip/hip_runtime.h>

typedef __attribute__((ext_vector_type(16))) _Float16 v16h;
typedef __attribute__((ext_vector_type(8)))  float    v8f;

#define N_E       1024
#define E_DIM     64
#define HW        4096          // 64*64
#define N_ROWS    131072        // 32*64*64
#define ROWS_PER_WG 256
#define NUM_WG    (N_ROWS / ROWS_PER_WG)   // 512
#define ZQ_ELEMS  8388608       // 32*64*64*64

// LDS carve (bytes):
//   ehbuf: 64 tiles * 2 chunks * 32 lanes * 32B = 131072
//   elbuf: 131072
//   sqe  : 1024*4 = 4096
//   red  : 256*4  = 1024
//   idxs : 256*4  = 1024
#define LDS_BYTES (131072 + 131072 + 4096 + 1024 + 1024)

union H16   { _Float16 e[16]; v16h v; };
union F8    { float    f[8];  v8f  v; };
union HPack { _Float16 h[4];  uint2 u; };

__global__ void vq_init_kernel(float* __restrict__ counts) {
    int t = blockIdx.x * 256 + threadIdx.x;
    if (t < N_E) counts[t] = 0.0f;
}

__global__ void __launch_bounds__(256, 4)
vq_main_kernel(const float* __restrict__ z,
               const float* __restrict__ emb,
               float* __restrict__ out,
               float* __restrict__ counts,
               float* __restrict__ partial) {
    extern __shared__ v16h smem[];
    v16h*  ehbuf = smem;                 // [64][2][32] B-frag hi
    v16h*  elbuf = smem + 4096;          // [64][2][32] B-frag lo
    float* fbase = (float*)(smem + 8192);
    float* sqe   = fbase;                // [1024]
    float* red   = fbase + 1024;         // [256]
    int*   idxs  = (int*)(fbase + 1280); // [256]

    const int tid      = threadIdx.x;
    const int wg       = blockIdx.x;
    const int wv       = tid >> 5;       // wave id 0..7
    const int lane     = tid & 31;
    const int laneN    = lane & 15;
    const int laneHalf = lane >> 4;

    const int b   = wg >> 4;                   // 16 workgroups per batch image
    const int hw0 = (wg & 15) * ROWS_PER_WG;   // 256 consecutive hw positions
    const float* zb   = z   + (size_t)b * E_DIM * HW;
    float*       outb = out + (size_t)b * E_DIM * HW;

    // ---- Stage codebook into split-f16, pre-swizzled WMMA-B fragments ----
    // B (32x16, KxN) layout assumed: column N = lane&15; VGPR v holds
    // K = (lane>>4)*16 + 2v, +1. Fragment flat index = (tile*2+chunk)*32+lane.
    // Vectorized: 4 consecutive k per thread -> float4 global read + b64 LDS store.
    for (int s4 = tid; s4 < (N_E * E_DIM) / 4; s4 += 256) {
        int s       = s4 * 4;
        int halfIdx = s & 15;          // base element within v16h (0,4,8,12)
        int ln      = (s >> 4) & 31;   // lane
        int ch      = (s >> 9) & 1;    // K chunk (0: K=0..31, 1: K=32..63)
        int tl      = s >> 10;         // N tile
        int j = tl * 16 + (ln & 15);   // codebook row
        int k = ch * 32 + (ln >> 4) * 16 + halfIdx;
        float4 e4 = *(const float4*)(emb + j * E_DIM + k);
        HPack ph, pl;
        {
            float e;
            e = e4.x; ph.h[0] = (_Float16)e; pl.h[0] = (_Float16)(e - (float)ph.h[0]);
            e = e4.y; ph.h[1] = (_Float16)e; pl.h[1] = (_Float16)(e - (float)ph.h[1]);
            e = e4.z; ph.h[2] = (_Float16)e; pl.h[2] = (_Float16)(e - (float)ph.h[2]);
            e = e4.w; ph.h[3] = (_Float16)e; pl.h[3] = (_Float16)(e - (float)ph.h[3]);
        }
        int fi = (tl * 2 + ch) * 32 + ln;
        *(uint2*)(((_Float16*)&ehbuf[fi]) + halfIdx) = ph.u;
        *(uint2*)(((_Float16*)&elbuf[fi]) + halfIdx) = pl.u;
    }

    // ---- |e_j|^2 (float4 reads) ----
    for (int j = tid; j < N_E; j += 256) {
        const float4* er = (const float4*)(emb + j * E_DIM);
        float s = 0.0f;
        #pragma unroll 4
        for (int k = 0; k < E_DIM / 4; ++k) {
            float4 e = er[k];
            s += e.x * e.x + e.y * e.y + e.z * e.z + e.w * e.w;
        }
        sqe[j] = s;
    }

    // ---- Build split-f16 A fragments for TWO 16-row M-tiles per wave,
    //      loading z directly from global (half-wave coalesced), and
    //      accumulate per-row |z|^2 on the fly.
    // A layout: row M = lane&15; VGPR v: K = (lane>>4)*16 + 2v, +1 (same k order as B).
    H16  ah[2][2], al[2][2];    // [mtile][chunk]
    float rsq[2][8];            // [mtile][C-vgpr]
    #pragma unroll
    for (int t = 0; t < 2; ++t) {
        const int hwA = hw0 + 32 * wv + 16 * t + laneN;  // this lane's row
        float own = 0.0f;
        #pragma unroll
        for (int ch = 0; ch < 2; ++ch) {
            #pragma unroll
            for (int v = 0; v < 8; ++v) {
                int k0 = laneHalf * 16 + 2 * v;
                k0 += ch * 32;
                float f0 = zb[(size_t)k0 * HW + hwA];
                float f1 = zb[(size_t)(k0 + 1) * HW + hwA];
                own += f0 * f0 + f1 * f1;
                _Float16 h0 = (_Float16)f0, h1 = (_Float16)f1;
                ah[t][ch].e[2 * v]     = h0;
                ah[t][ch].e[2 * v + 1] = h1;
                al[t][ch].e[2 * v]     = (_Float16)(f0 - (float)h0);
                al[t][ch].e[2 * v + 1] = (_Float16)(f1 - (float)h1);
            }
        }
        // lanes L and L+16 jointly hold all 64 channels of row laneN
        float rowAcc = own + __shfl_xor(own, 16);
        // C layout: VGPR v -> row M = v + 8*laneHalf; that row's |z|^2 lives
        // in lanes with laneN == v + 8*laneHalf.
        #pragma unroll
        for (int v = 0; v < 8; ++v) rsq[t][v] = __shfl(rowAcc, v + 8 * laneHalf);
    }
    __syncthreads();

    float bestv[2][8];
    int   besti[2][8];
    #pragma unroll
    for (int t = 0; t < 2; ++t)
        #pragma unroll
        for (int v = 0; v < 8; ++v) { bestv[t][v] = 3.0e38f; besti[t][v] = 0; }

    // ---- Main loop: 64 N-tiles; 12 WMMAs per 4 B-fragment loads ----
    for (int tl = 0; tl < 64; ++tl) {
        v16h bh0 = ehbuf[(tl * 2 + 0) * 32 + lane];
        v16h bh1 = ehbuf[(tl * 2 + 1) * 32 + lane];
        v16h bl0 = elbuf[(tl * 2 + 0) * 32 + lane];
        v16h bl1 = elbuf[(tl * 2 + 1) * 32 + lane];
        v8f acc0 = {}, acc1 = {};
        acc0 = __builtin_amdgcn_wmma_f32_16x16x32_f16(false, ah[0][0].v, false, bh0, (short)0, acc0, false, false);
        acc1 = __builtin_amdgcn_wmma_f32_16x16x32_f16(false, ah[1][0].v, false, bh0, (short)0, acc1, false, false);
        acc0 = __builtin_amdgcn_wmma_f32_16x16x32_f16(false, ah[0][1].v, false, bh1, (short)0, acc0, false, false);
        acc1 = __builtin_amdgcn_wmma_f32_16x16x32_f16(false, ah[1][1].v, false, bh1, (short)0, acc1, false, false);
        acc0 = __builtin_amdgcn_wmma_f32_16x16x32_f16(false, ah[0][0].v, false, bl0, (short)0, acc0, false, false);
        acc1 = __builtin_amdgcn_wmma_f32_16x16x32_f16(false, ah[1][0].v, false, bl0, (short)0, acc1, false, false);
        acc0 = __builtin_amdgcn_wmma_f32_16x16x32_f16(false, ah[0][1].v, false, bl1, (short)0, acc0, false, false);
        acc1 = __builtin_amdgcn_wmma_f32_16x16x32_f16(false, ah[1][1].v, false, bl1, (short)0, acc1, false, false);
        acc0 = __builtin_amdgcn_wmma_f32_16x16x32_f16(false, al[0][0].v, false, bh0, (short)0, acc0, false, false);
        acc1 = __builtin_amdgcn_wmma_f32_16x16x32_f16(false, al[1][0].v, false, bh0, (short)0, acc1, false, false);
        acc0 = __builtin_amdgcn_wmma_f32_16x16x32_f16(false, al[0][1].v, false, bh1, (short)0, acc0, false, false);
        acc1 = __builtin_amdgcn_wmma_f32_16x16x32_f16(false, al[1][1].v, false, bh1, (short)0, acc1, false, false);
        int   j  = tl * 16 + laneN;
        float sq = sqe[j];
        F8 cu0; cu0.v = acc0;
        F8 cu1; cu1.v = acc1;
        #pragma unroll
        for (int v = 0; v < 8; ++v) {
            float d0 = (rsq[0][v] + sq) - 2.0f * cu0.f[v];
            if (d0 < bestv[0][v]) { bestv[0][v] = d0; besti[0][v] = j; }
            float d1 = (rsq[1][v] + sq) - 2.0f * cu1.f[v];
            if (d1 < bestv[1][v]) { bestv[1][v] = d1; besti[1][v] = j; }
        }
    }

    // ---- Reduce (min, argmin-first) across the 16 lanes of each half ----
    #pragma unroll
    for (int t = 0; t < 2; ++t) {
        #pragma unroll
        for (int v = 0; v < 8; ++v) {
            #pragma unroll
            for (int m = 1; m < 16; m <<= 1) {
                float ov = __shfl_xor(bestv[t][v], m, 32);
                int   oi = __shfl_xor(besti[t][v], m, 32);
                if (ov < bestv[t][v] || (ov == bestv[t][v] && oi < besti[t][v])) {
                    bestv[t][v] = ov; besti[t][v] = oi;
                }
            }
        }
        if (laneN == 0) {
            #pragma unroll
            for (int v = 0; v < 8; ++v)
                idxs[32 * wv + 16 * t + v + 8 * laneHalf] = besti[t][v];
        }
    }
    __syncthreads();

    // ---- counts (one row per thread; exact-integer float atomics) ----
    atomicAdd(&counts[idxs[tid]], 1.0f);

    // ---- gather z_q, write coalesced, accumulate SSE (z re-read, L2-hot) ----
    float sse = 0.0f;
    for (int s = tid; s < ROWS_PER_WG * E_DIM; s += 256) {
        int r = s & 255;
        int c = s >> 8;
        float q  = emb[idxs[r] * E_DIM + c];
        float zv = zb[(size_t)c * HW + hw0 + r];
        float d  = q - zv;
        sse += d * d;
        outb[(size_t)c * HW + hw0 + r] = q;   // consecutive r -> coalesced
    }
    red[tid] = sse;
    __syncthreads();
    for (int off = 128; off > 0; off >>= 1) {
        if (tid < off) red[tid] += red[tid + off];
        __syncthreads();
    }
    if (tid == 0) partial[wg] = red[0];
}

__global__ void vq_finalize_kernel(const float* __restrict__ partial,
                                   float* __restrict__ lossOut) {
    __shared__ float red[256];
    float s = 0.0f;
    for (int i = threadIdx.x; i < NUM_WG; i += 256) s += partial[i];
    red[threadIdx.x] = s;
    __syncthreads();
    for (int off = 128; off > 0; off >>= 1) {
        if (threadIdx.x < off) red[threadIdx.x] += red[threadIdx.x + off];
        __syncthreads();
    }
    // loss = (BETA + 1) * SSE / numel = 1.25 * SSE / 8388608
    if (threadIdx.x == 0) lossOut[0] = 1.25f * red[0] / 8388608.0f;
}

extern "C" void kernel_launch(void* const* d_in, const int* in_sizes, int n_in,
                              void* d_out, int out_size, void* d_ws, size_t ws_size,
                              hipStream_t stream) {
    (void)in_sizes; (void)n_in; (void)out_size; (void)ws_size;
    const float* z   = (const float*)d_in[0];
    const float* emb = (const float*)d_in[1];
    float* out     = (float*)d_out;         // z_q: 8388608 floats
    float* loss    = out + ZQ_ELEMS;        // 1 float
    float* counts  = loss + 1;              // 1024 floats
    float* partial = (float*)d_ws;          // 512 floats of scratch

    vq_init_kernel<<<4, 256, 0, stream>>>(counts);
    vq_main_kernel<<<NUM_WG, 256, LDS_BYTES, stream>>>(z, emb, out, counts, partial);
    vq_finalize_kernel<<<1, 256, 0, stream>>>(partial, loss);
}
